// MySpikeGPT_35948876268214
// MI455X (gfx1250) — compile-verified, matrix operand order checked by reading, not run on hardware
//
#include <hip/hip_runtime.h>

// ---------------------------------------------------------------------------
// MySpikeGPT forward on MI455X (gfx1250): all GEMMs via v_wmma_f32_16x16x32_bf16
// ---------------------------------------------------------------------------
#define RTOT 2048     // B*S rows
#define SEQ  1024
#define DEMB 768
#define NH   12
#define HD   64
#define FFNH 3072
#define NVOCAB 32000

#define SA_STRIDE 40  // bf16 elements per LDS tile row (16B-aligned row starts)

typedef __attribute__((ext_vector_type(16))) __bf16 v16bf;
typedef __attribute__((ext_vector_type(8)))  float  v8f;

__device__ __forceinline__ v8f zero8() {
  v8f z;
#pragma unroll
  for (int i = 0; i < 8; ++i) z[i] = 0.0f;
  return z;
}

// qrelu with precomputed reciprocal: clip(relu(x)*ia,0,1) -> round(*15)/15 -> *a
__device__ __forceinline__ float qrelu_fast(float x, float alpha, float inv_alpha) {
  float xc = fminf(fmaxf(x, 0.0f) * inv_alpha, 1.0f);
  return rintf(xc * 15.0f) * (1.0f / 15.0f) * alpha;
}

// pack two f32 -> packed bf16 dword
__device__ __forceinline__ unsigned pk2(float x, float y) {
  union { __bf16 b[2]; unsigned u; } u;
  u.b[0] = (__bf16)x; u.b[1] = (__bf16)y;
  return u.u;
}

// K index for 16-bit A/B fragment, dword-pair p (0..7), lane half
__device__ __forceinline__ int kmap(int p, int lh) {
  return (p < 4) ? (lh * 8 + 2 * p) : (16 + lh * 8 + 2 * (p - 4));
}

// A-style fragment from bf16 LDS tile (rows indexed by lane&15, cols by kmap)
__device__ __forceinline__ v16bf frag_from_lds(const __bf16* base, int stride, int lane) {
  const int r = lane & 15, lh = (lane >> 4) & 1;
  const __bf16* row = base + r * stride;
  v16bf out;
#pragma unroll
  for (int p = 0; p < 8; ++p) {
    int k = kmap(p, lh);
    out[2 * p]     = row[k];
    out[2 * p + 1] = row[k + 1];
  }
  return out;
}

// A-style fragment from f32 memory (rows by lane&15, elements walk K in kmap order)
__device__ __forceinline__ v16bf afrag_from_g(const float* base, int rowStride, int lane) {
  const int r = lane & 15, lh = (lane >> 4) & 1;
  const float* row = base + (size_t)r * rowStride;
  v16bf out;
#pragma unroll
  for (int p = 0; p < 8; ++p) {
    int k = kmap(p, lh);
    out[2 * p]     = (__bf16)row[k];
    out[2 * p + 1] = (__bf16)row[k + 1];
  }
  return out;
}

// B-style fragment from f32 memory (column n = lane&15 fixed, elements walk K rows)
__device__ __forceinline__ v16bf bfrag_from_g(const float* base, int rowStride, int lane) {
  const int n = lane & 15, lh = (lane >> 4) & 1;
  v16bf out;
#pragma unroll
  for (int p = 0; p < 8; ++p) {
    int k = kmap(p, lh);
    out[2 * p]     = (__bf16)base[(size_t)k * rowStride + n];
    out[2 * p + 1] = (__bf16)base[(size_t)(k + 1) * rowStride + n];
  }
  return out;
}

__device__ __forceinline__ v8f wmma_bf16(v16bf a, v16bf b, v8f c) {
  return __builtin_amdgcn_wmma_f32_16x16x32_bf16(false, a, false, b, (short)0, c, false, false);
}

__device__ __forceinline__ float redmax16(float v) {
  v = fmaxf(v, __shfl_xor(v, 1, 32));
  v = fmaxf(v, __shfl_xor(v, 2, 32));
  v = fmaxf(v, __shfl_xor(v, 4, 32));
  v = fmaxf(v, __shfl_xor(v, 8, 32));
  return v;
}
__device__ __forceinline__ float redsum16(float v) {
  v += __shfl_xor(v, 1, 32);
  v += __shfl_xor(v, 2, 32);
  v += __shfl_xor(v, 4, 32);
  v += __shfl_xor(v, 8, 32);
  return v;
}

// ---------------------------------------------------------------------------
// Embedding gather + positional add + qrelu(alpha_init)
// ---------------------------------------------------------------------------
__global__ __launch_bounds__(256) void embed_kernel(
    const int* __restrict__ x, const float* __restrict__ emb,
    const float* __restrict__ pos, const float* __restrict__ alphaPtr,
    float* __restrict__ X) {
  const int row = blockIdx.x, tid = threadIdx.x;
  const int tok = x[row];
  const int s = row & (SEQ - 1);
  const float alpha = alphaPtr[0];
  const float ia = 1.0f / alpha;
#pragma unroll
  for (int j = 0; j < 3; ++j) {
    int d = tid + 256 * j;
    float v = emb[(size_t)tok * DEMB + d] + pos[(size_t)s * DEMB + d];
    X[(size_t)row * DEMB + d] = qrelu_fast(v, alpha, ia);
  }
}

// ---------------------------------------------------------------------------
// LayerNorm (biased variance, eps=1e-5), one row per block
// ---------------------------------------------------------------------------
__global__ __launch_bounds__(256) void ln_kernel(
    const float* __restrict__ X, const float* __restrict__ w,
    const float* __restrict__ b, float* __restrict__ Out) {
  __shared__ float red[256];
  __shared__ float red2[256];
  const int row = blockIdx.x, tid = threadIdx.x;
  const float* xr = X + (size_t)row * DEMB;
  float v[3], s = 0.f, s2 = 0.f;
#pragma unroll
  for (int j = 0; j < 3; ++j) {
    v[j] = xr[tid + 256 * j];
    s += v[j];
    s2 += v[j] * v[j];
  }
  red[tid] = s; red2[tid] = s2;
  __syncthreads();
  for (int off = 128; off > 0; off >>= 1) {
    if (tid < off) { red[tid] += red[tid + off]; red2[tid] += red2[tid + off]; }
    __syncthreads();
  }
  const float mean = red[0] * (1.0f / DEMB);
  const float var  = red2[0] * (1.0f / DEMB) - mean * mean;
  const float inv  = rsqrtf(var + 1e-5f);
#pragma unroll
  for (int j = 0; j < 3; ++j) {
    int d = tid + 256 * j;
    Out[(size_t)row * DEMB + d] = (v[j] - mean) * inv * w[d] + b[d];
  }
}

// ---------------------------------------------------------------------------
// GEMM: Out[M,N] = (optional Residual +) qrelu(A[M,K] @ W[K,N], alpha)
// 128x128 tile / block (8 waves), K-step 32, WMMA bf16, vectorized LDS staging.
// ---------------------------------------------------------------------------
__global__ __launch_bounds__(256) void gemm_qrelu_kernel(
    const float* __restrict__ A, const float* __restrict__ W,
    float* __restrict__ Out, const float* __restrict__ Residual,
    const float* __restrict__ alphaPtr, int M, int N, int K) {
  __shared__ __bf16 sA[128 * SA_STRIDE];   // [m][k]
  __shared__ __bf16 sB[128 * SA_STRIDE];   // transposed: [n][k]
  const int tid = threadIdx.x;
  const int lane = tid & 31;
  const int wave = tid >> 5;
  const int wm = wave & 1;          // 0..1 -> 64-row halves
  const int wn = wave >> 1;         // 0..3 -> 32-col quarters
  const int mBlock = blockIdx.x * 128;
  const int nBlock = blockIdx.y * 128;

  // A-tile staging map: each thread owns 16 contiguous elems of one row
  const int ar = tid >> 1, ac = (tid & 1) * 16;
  // W-tile staging map: each thread owns 16 contiguous cols of one k-row
  const int wk = tid >> 3, wc = (tid & 7) * 16;

  v8f acc[4][2];
#pragma unroll
  for (int mi = 0; mi < 4; ++mi)
#pragma unroll
    for (int ni = 0; ni < 2; ++ni) acc[mi][ni] = zero8();

  const int nK = K >> 5;
  for (int kt = 0; kt < nK; ++kt) {
    const int k0 = kt * 32;
    // ---- A tile 128x32: 4x global_load_b128, 2x ds_store_b128 (packed bf16)
    {
      const float4* src = reinterpret_cast<const float4*>(
          A + (size_t)(mBlock + ar) * K + k0 + ac);
      float4 f0 = src[0], f1 = src[1], f2 = src[2], f3 = src[3];
      uint4* dst = reinterpret_cast<uint4*>(&sA[ar * SA_STRIDE + ac]);
      dst[0] = make_uint4(pk2(f0.x, f0.y), pk2(f0.z, f0.w),
                          pk2(f1.x, f1.y), pk2(f1.z, f1.w));
      dst[1] = make_uint4(pk2(f2.x, f2.y), pk2(f2.z, f2.w),
                          pk2(f3.x, f3.y), pk2(f3.z, f3.w));
    }
    // ---- W tile 32x128: 4x global_load_b128, transposed scalar stores
    {
      const float4* src = reinterpret_cast<const float4*>(
          W + (size_t)(k0 + wk) * N + nBlock + wc);
      float4 f[4];
      f[0] = src[0]; f[1] = src[1]; f[2] = src[2]; f[3] = src[3];
#pragma unroll
      for (int q = 0; q < 4; ++q) {
        sB[(wc + 4 * q + 0) * SA_STRIDE + wk] = (__bf16)f[q].x;
        sB[(wc + 4 * q + 1) * SA_STRIDE + wk] = (__bf16)f[q].y;
        sB[(wc + 4 * q + 2) * SA_STRIDE + wk] = (__bf16)f[q].z;
        sB[(wc + 4 * q + 3) * SA_STRIDE + wk] = (__bf16)f[q].w;
      }
    }
    __syncthreads();
    v16bf af[4], bfr[2];
#pragma unroll
    for (int mi = 0; mi < 4; ++mi)
      af[mi] = frag_from_lds(sA + (wm * 64 + mi * 16) * SA_STRIDE, SA_STRIDE, lane);
#pragma unroll
    for (int ni = 0; ni < 2; ++ni)
      bfr[ni] = frag_from_lds(sB + (wn * 32 + ni * 16) * SA_STRIDE, SA_STRIDE, lane);
#pragma unroll
    for (int mi = 0; mi < 4; ++mi)
#pragma unroll
      for (int ni = 0; ni < 2; ++ni)
        acc[mi][ni] = wmma_bf16(af[mi], bfr[ni], acc[mi][ni]);
    __syncthreads();
  }

  const float alpha = alphaPtr[0];
  const float ia = 1.0f / alpha;
  const int lh = lane >> 4, nl = lane & 15;
  const bool hasRes = (Residual != nullptr);
#pragma unroll
  for (int mi = 0; mi < 4; ++mi)
#pragma unroll
    for (int ni = 0; ni < 2; ++ni)
#pragma unroll
      for (int i = 0; i < 8; ++i) {
        int row = mBlock + wm * 64 + mi * 16 + i + lh * 8;
        int col = nBlock + wn * 32 + ni * 16 + nl;
        float v = qrelu_fast(acc[mi][ni][i], alpha, ia);
        if (hasRes) v += Residual[(size_t)row * N + col];
        Out[(size_t)row * N + col] = v;
      }
}

// ---------------------------------------------------------------------------
// Causal attention, two-pass flash per 16-query tile, one wave per block.
// scores = QK^T / sqrt(768); softmax; exact qrelu(a_sm) on probs; O=P@V;
// qrelu(a_w) on output.  QK^T and P@V both on WMMA bf16.
// ---------------------------------------------------------------------------
__global__ __launch_bounds__(32) void attn_kernel(
    const float* __restrict__ Q, const float* __restrict__ K,
    const float* __restrict__ V, float* __restrict__ O,
    const float* __restrict__ aSm, const float* __restrict__ aW) {
  __shared__ float sP[16 * 32];
  const int lane = threadIdx.x;
  const int bid = blockIdx.x;
  const int qt = bid & 63;
  const int h  = (bid >> 6) % NH;
  const int b  = bid / (64 * NH);
  const int q0 = qt * 16;
  const float scale = 0.03608439182435161f;   // 1/sqrt(768) (scaled by embed dim, as in ref)
  const float alpha_sm = aSm[0], alpha_w = aW[0];
  const float ia_sm = 1.0f / alpha_sm, ia_w = 1.0f / alpha_w;
  const int lh = lane >> 4, nl = lane & 15;

  const size_t baseQ = ((size_t)b * SEQ + q0) * DEMB + h * HD;
  v16bf qf[2];
  qf[0] = afrag_from_g(Q + baseQ, DEMB, lane);
  qf[1] = afrag_from_g(Q + baseQ + 32, DEMB, lane);

  float mrun[8], lrun[8];
#pragma unroll
  for (int i = 0; i < 8; ++i) { mrun[i] = -1e30f; lrun[i] = 0.0f; }

  const int nChunks = (q0 + 16 + 31) >> 5;

  // ---- pass 1: streaming row max / sum of exp ----
  for (int c = 0; c < nChunks; ++c) {
    const int kc0 = c * 32;
    v8f s[2]; s[0] = zero8(); s[1] = zero8();
#pragma unroll
    for (int t = 0; t < 2; ++t)
#pragma unroll
      for (int j = 0; j < 2; ++j) {
        const size_t bk = ((size_t)b * SEQ + kc0 + 16 * t) * DEMB + h * HD + j * 32;
        s[t] = wmma_bf16(qf[j], afrag_from_g(K + bk, DEMB, lane), s[t]);
      }
#pragma unroll
    for (int i = 0; i < 8; ++i) {
      const int qrow = q0 + i + lh * 8;
      float sv0 = s[0][i] * scale, sv1 = s[1][i] * scale;
      if (kc0 + nl > qrow)      sv0 = -1e30f;
      if (kc0 + 16 + nl > qrow) sv1 = -1e30f;
      float mx = redmax16(fmaxf(sv0, sv1));
      float mnew = fmaxf(mrun[i], mx);
      float ps = redsum16(__expf(sv0 - mnew) + __expf(sv1 - mnew));
      lrun[i] = lrun[i] * __expf(mrun[i] - mnew) + ps;
      mrun[i] = mnew;
    }
  }

  float invl[8];
#pragma unroll
  for (int i = 0; i < 8; ++i) invl[i] = 1.0f / lrun[i];

  // ---- pass 2: probabilities (exact quantization) and O = P @ V ----
  v8f oacc[4];
#pragma unroll
  for (int t = 0; t < 4; ++t) oacc[t] = zero8();

  for (int c = 0; c < nChunks; ++c) {
    const int kc0 = c * 32;
    v8f s[2]; s[0] = zero8(); s[1] = zero8();
#pragma unroll
    for (int t = 0; t < 2; ++t)
#pragma unroll
      for (int j = 0; j < 2; ++j) {
        const size_t bk = ((size_t)b * SEQ + kc0 + 16 * t) * DEMB + h * HD + j * 32;
        s[t] = wmma_bf16(qf[j], afrag_from_g(K + bk, DEMB, lane), s[t]);
      }
#pragma unroll
    for (int t = 0; t < 2; ++t)
#pragma unroll
      for (int i = 0; i < 8; ++i) {
        const int qrow = q0 + i + lh * 8;
        float sv = s[t][i] * scale;
        if (kc0 + 16 * t + nl > qrow) sv = -1e30f;
        float p = __expf(sv - mrun[i]) * invl[i];
        sP[(i + 8 * lh) * 32 + 16 * t + nl] = qrelu_fast(p, alpha_sm, ia_sm);
      }
    __syncthreads();   // single-wave WG: S_NOP barrier + mem ordering fence
    v16bf pf = afrag_from_g(sP, 32, lane);
#pragma unroll
    for (int t = 0; t < 4; ++t) {
      const size_t bv = ((size_t)b * SEQ + kc0) * DEMB + h * HD + 16 * t;
      oacc[t] = wmma_bf16(pf, bfrag_from_g(V + bv, DEMB, lane), oacc[t]);
    }
    __syncthreads();   // WAR on sP before next chunk
  }

#pragma unroll
  for (int t = 0; t < 4; ++t)
#pragma unroll
    for (int i = 0; i < 8; ++i) {
      int row = b * SEQ + q0 + i + lh * 8;
      int col = h * HD + 16 * t + nl;
      O[(size_t)row * DEMB + col] = qrelu_fast(oacc[t][i], alpha_w, ia_w);
    }
}

// ---------------------------------------------------------------------------
// Last-token logits: out[b, v] = X[row_b, :] . out_w[:, v]
// ---------------------------------------------------------------------------
__global__ __launch_bounds__(256) void logits_kernel(
    const float* __restrict__ X, const float* __restrict__ W,
    float* __restrict__ out) {
  __shared__ float xr[DEMB];
  const int bb = blockIdx.y, tid = threadIdx.x;
  const size_t row = (size_t)bb * SEQ + (SEQ - 1);
#pragma unroll
  for (int j = 0; j < 3; ++j) xr[tid + 256 * j] = X[row * DEMB + tid + 256 * j];
  __syncthreads();
  const int v = blockIdx.x * 256 + tid;
  float acc = 0.0f;
  for (int d = 0; d < DEMB; ++d) acc += xr[d] * W[(size_t)d * NVOCAB + v];
  out[(size_t)bb * NVOCAB + v] = acc;
}

// ---------------------------------------------------------------------------
// Launch
// ---------------------------------------------------------------------------
extern "C" void kernel_launch(void* const* d_in, const int* in_sizes, int n_in,
                              void* d_out, int out_size, void* d_ws, size_t ws_size,
                              hipStream_t stream) {
  (void)in_sizes; (void)n_in; (void)out_size; (void)ws_size;
  const int*   x      = (const int*)d_in[0];
  const float* emb    = (const float*)d_in[1];
  const float* pos    = (const float*)d_in[2];
  const float* out_w  = (const float*)d_in[3];
  const float* alpha0 = (const float*)d_in[4];

  float* ws = (float*)d_ws;
  const size_t RD = (size_t)RTOT * DEMB;        // 1.57M floats
  float* X  = ws;                               // activations (residual stream)
  float* T  = X  + RD;                          // layernorm output
  float* Qb = T  + RD;
  float* Kb = Qb + RD;
  float* Vb = Kb + RD;
  float* Ob = Vb + RD;
  float* H1 = Ob + RD;                          // 2048 x 3072 FFN hidden

  embed_kernel<<<RTOT, 256, 0, stream>>>(x, emb, pos, alpha0, X);

  const dim3 g768(16, 6), g3072(16, 24);
  for (int L = 0; L < 4; ++L) {
    const int base = 5 + 18 * L;
    const float* wq   = (const float*)d_in[base + 0];
    const float* wk   = (const float*)d_in[base + 1];
    const float* wv   = (const float*)d_in[base + 2];
    const float* wo   = (const float*)d_in[base + 3];
    const float* w1   = (const float*)d_in[base + 4];
    const float* w2   = (const float*)d_in[base + 5];
    const float* ln1w = (const float*)d_in[base + 6];
    const float* ln1b = (const float*)d_in[base + 7];
    const float* ln2w = (const float*)d_in[base + 8];
    const float* ln2b = (const float*)d_in[base + 9];
    const float* a_q  = (const float*)d_in[base + 10];
    const float* a_k  = (const float*)d_in[base + 11];
    const float* a_v  = (const float*)d_in[base + 12];
    const float* a_sm = (const float*)d_in[base + 13];
    const float* a_w  = (const float*)d_in[base + 14];
    const float* a_o  = (const float*)d_in[base + 15];
    const float* a_1  = (const float*)d_in[base + 16];
    const float* a_2  = (const float*)d_in[base + 17];

    ln_kernel<<<RTOT, 256, 0, stream>>>(X, ln1w, ln1b, T);
    gemm_qrelu_kernel<<<g768, 256, 0, stream>>>(T, wq, Qb, nullptr, a_q, RTOT, DEMB, DEMB);
    gemm_qrelu_kernel<<<g768, 256, 0, stream>>>(T, wk, Kb, nullptr, a_k, RTOT, DEMB, DEMB);
    gemm_qrelu_kernel<<<g768, 256, 0, stream>>>(T, wv, Vb, nullptr, a_v, RTOT, DEMB, DEMB);
    attn_kernel<<<2 * NH * 64, 32, 0, stream>>>(Qb, Kb, Vb, Ob, a_sm, a_w);
    gemm_qrelu_kernel<<<g768, 256, 0, stream>>>(Ob, wo, X, X, a_o, RTOT, DEMB, DEMB);
    ln_kernel<<<RTOT, 256, 0, stream>>>(X, ln2w, ln2b, T);
    gemm_qrelu_kernel<<<g3072, 256, 0, stream>>>(T, w1, H1, nullptr, a_1, RTOT, FFNH, DEMB);
    gemm_qrelu_kernel<<<g768, 256, 0, stream>>>(H1, w2, X, X, a_2, RTOT, DEMB, FFNH);
  }

  logits_kernel<<<dim3(NVOCAB / 256, 2), 256, 0, stream>>>(X, out_w, (float*)d_out);
}
